// GliZNetLoss_3478923509848
// MI455X (gfx1250) — matrix operand order repaired and checked by reading, not compiled, not run on hardware
//
#include <hip/hip_runtime.h>
#include <hip/hip_bf16.h>
#include <math.h>
#include <stdint.h>

typedef __bf16 v16bf __attribute__((ext_vector_type(16)));
typedef __bf16 v8bf  __attribute__((ext_vector_type(8)));
typedef float  v8f   __attribute__((ext_vector_type(8)));
typedef int    v4i   __attribute__((ext_vector_type(4)));
typedef v4i __attribute__((address_space(1))) as1_v4i;   // global-space int4
typedef v4i __attribute__((address_space(3))) as3_v4i;   // LDS-space int4

#define BB 256
#define LL 32
#define DD 768
#define FSTR 772               // f32 LDS row stride (3088 B, bank stride 4)
#define HSTR 776               // bf16 LDS row stride (1552 B, 16B aligned, bank stride 4)
#define NEG_INF (-1e30f)
#define REP_THR 0.3f

// dynamic-LDS partition (bytes)
#define EF_OFF    0
#define EBF_OFF   (LL * FSTR * 4)              // 98816
#define LBL_OFF   (EBF_OFF + LL * HSTR * 2)    // 148480
#define BAT_OFF   (LBL_OFF + LL * 8)           // 148736
#define PART_OFF  (BAT_OFF + LL * 8)           // 148992
#define INV_OFF   (PART_OFF + LL * 4 * 4)      // 149504
#define WSUM_OFF  (INV_OFF + LL * 4)           // 149632
#define WCNT_OFF  (WSUM_OFF + 16)              // 149648
#define SMEM_BYTES (WCNT_OFF + 16)             // 149664 (< 320KB/WGP)

// ---- gfx1250 async global->LDS copy (16 bytes per lane), ASYNCcnt-tracked ----
__device__ __forceinline__ void async_copy_b128(const void* gsrc, void* ldst) {
#if __has_builtin(__builtin_amdgcn_global_load_async_to_lds_b128)
    __builtin_amdgcn_global_load_async_to_lds_b128(
        (as1_v4i*)(uintptr_t)gsrc,
        (as3_v4i*)(uint32_t)(uintptr_t)ldst,
        0, 0);
#else
    unsigned lds_addr = (unsigned)(uintptr_t)ldst;
    asm volatile("global_load_async_to_lds_b128 %0, %1, off"
                 :: "v"(lds_addr), "v"(gsrc) : "memory");
#endif
}

__device__ __forceinline__ void wait_async_zero() {
#if __has_builtin(__builtin_amdgcn_s_wait_asynccnt)
    __builtin_amdgcn_s_wait_asynccnt(0);
#else
    asm volatile("s_wait_asynccnt 0" ::: "memory");
#endif
}

// One block (128 threads = 4 wave32) per sample b.
__global__ __launch_bounds__(128)
void gliz_block_kernel(const float* __restrict__ logits,        // (N,1)
                       const int* __restrict__ labels,          // (B,L) int32
                       const long long* __restrict__ batch_idx, // (N) int64
                       const long long* __restrict__ label_ids, // (N) int64
                       const float* __restrict__ emb,           // (N,D)
                       const float* __restrict__ logit_scale,   // (1)
                       const float* __restrict__ bce_scale,     // (1)
                       float* __restrict__ ws_rep_sum,
                       float* __restrict__ ws_sup_sum,
                       float* __restrict__ ws_bce_sum,
                       unsigned* __restrict__ ws_rep_cnt,
                       unsigned* __restrict__ ws_anchor,
                       unsigned* __restrict__ ws_bce_cnt)
{
    extern __shared__ char smem[];
    float*     Ef   = (float*)(smem + EF_OFF);      // f32 staging, stride FSTR
    __bf16*    Ebf  = (__bf16*)(smem + EBF_OFF);    // normalized bf16, stride HSTR
    long long* LBL  = (long long*)(smem + LBL_OFF);
    long long* BAT  = (long long*)(smem + BAT_OFF);
    float (*partial)[4] = (float (*)[4])(smem + PART_OFF);
    float*     inv_n    = (float*)(smem + INV_OFF);
    float*     wave_sum = (float*)(smem + WSUM_OFF);
    unsigned*  wave_cnt = (unsigned*)(smem + WCNT_OFF);

    const int b = blockIdx.x;
    const int t = threadIdx.x;
    const float* Eb = emb + (size_t)b * LL * DD;

    // ---------- phase 0: async-stage 32x768 f32 tile into LDS (HBM read once) ----
    {
        const int rr = t >> 2;         // row 0..31
        const int qq = t & 3;          // quarter: 192 floats = 48 x 16B chunks
        const char* gsrc = (const char*)(Eb + rr * DD + qq * 192);
        char*       ldst = (char*)(Ef + rr * FSTR + qq * 192);
        #pragma unroll 8
        for (int i = 0; i < 48; ++i)
            async_copy_b128(gsrc + i * 16, ldst + i * 16);

        if (t < LL) {                  // stage per-block mask metadata
            LBL[t] = label_ids[b * LL + t];
            BAT[t] = batch_idx[b * LL + t];
        }
        wait_async_zero();
    }
    __syncthreads();

    // ---------- phase 1: per-row sum of squares (from LDS) ----------
    const int r = t & 31;              // row
    const int q = t >> 5;              // quarter
    {
        const float* p = Ef + r * FSTR + q * 192;
        float s = 0.f;
        #pragma unroll 8
        for (int i = 0; i < 192; i += 4) {
            float4 v = *(const float4*)(p + i);
            s += v.x * v.x + v.y * v.y + v.z * v.z + v.w * v.w;
        }
        partial[r][q] = s;
    }
    __syncthreads();
    if (t < LL) {
        float s = partial[t][0] + partial[t][1] + partial[t][2] + partial[t][3];
        inv_n[t] = 1.0f / fmaxf(sqrtf(s), 1e-12f);
    }
    __syncthreads();

    // ---------- phase 2: normalize + convert to bf16 (LDS -> LDS) ----------
    {
        const float inv = inv_n[r];
        const float* p = Ef + r * FSTR + q * 192;
        __bf16* o = Ebf + r * HSTR + q * 192;
        #pragma unroll 8
        for (int i = 0; i < 192; i += 4) {
            float4 v = *(const float4*)(p + i);
            o[i + 0] = (__bf16)(v.x * inv);
            o[i + 1] = (__bf16)(v.y * inv);
            o[i + 2] = (__bf16)(v.z * inv);
            o[i + 3] = (__bf16)(v.w * inv);
        }
    }
    __syncthreads();

    // ---------- phase 3: 32x32 Gram matrix via bf16 WMMA ----------
    // wave w computes 16x16 tile (ra, rb) of G = En * En^T
    const int w    = t >> 5;
    const int lane = t & 31;
    const int half = lane >> 4;
    const int m    = lane & 15;
    const int ra = (w >> 1) * 16;
    const int rb = (w & 1)  * 16;

    // A (16x32 bf16): lane half 0 holds K {0..7,16..23}, half 1 holds K {8..15,24..31}
    const __bf16* pa = Ebf + (ra + m) * HSTR + half * 8;
    // B (32x16 bf16), B[k][n] = E[rb+n][k]: half 0 holds K {0..7,16..23}, half 1 K {8..15,24..31}
    const __bf16* pb = Ebf + (rb + m) * HSTR + half * 8;

    v8f acc = {0.f, 0.f, 0.f, 0.f, 0.f, 0.f, 0.f, 0.f};
    #pragma unroll 4
    for (int k0 = 0; k0 < DD; k0 += 32) {
        v8bf alo = *(const v8bf*)(pa + k0);
        v8bf ahi = *(const v8bf*)(pa + k0 + 16);
        v8bf blo = *(const v8bf*)(pb + k0);
        v8bf bhi = *(const v8bf*)(pb + k0 + 16);
        v16bf av = __builtin_shufflevector(alo, ahi, 0,1,2,3,4,5,6,7,8,9,10,11,12,13,14,15);
        v16bf bv = __builtin_shufflevector(blo, bhi, 0,1,2,3,4,5,6,7,8,9,10,11,12,13,14,15);
        acc = __builtin_amdgcn_wmma_f32_16x16x32_bf16(
                  false, av, false, bv, (short)0, acc, false, false);
    }

    // C/D layout: VGPR rr -> M = rr + half*8, N = m; branchless masked penalty.
    {
        const long long lj = LBL[rb + m];
        const long long bj = BAT[rb + m];
        float psum = 0.f;
        unsigned pcnt = 0;
        #pragma unroll
        for (int rr = 0; rr < 8; ++rr) {
            const int ai = ra + rr + half * 8;
            const unsigned mk = (LBL[ai] != lj) & (BAT[ai] == bj);
            const float pm = fmaxf(acc[rr] - REP_THR, 0.f);
            psum = fmaf((float)mk, pm, psum);
            pcnt += mk;
        }
        for (int o = 16; o > 0; o >>= 1) {
            psum += __shfl_xor(psum, o, 32);
            pcnt += __shfl_xor(pcnt, o, 32);
        }
        if (lane == 0) { wave_sum[w] = psum; wave_cnt[w] = pcnt; }
    }
    __syncthreads();
    if (t == 0) {
        ws_rep_sum[b] = wave_sum[0] + wave_sum[1] + wave_sum[2] + wave_sum[3];
        ws_rep_cnt[b] = wave_cnt[0] + wave_cnt[1] + wave_cnt[2] + wave_cnt[3];
    }

    // ---------- phase 4: SupCon + BCE (wave 0, 32 lanes = 32 labels) ----------
    if (t < LL) {
        const float v  = logits[b * LL + t];          // dense_logits[b][t] (identity scatter)
        const float tl = (float)labels[b * LL + t];
        const bool valid = (tl != -100.0f);
        const float xm = valid ? v : NEG_INF;

        float mx = xm;
        for (int o = 16; o > 0; o >>= 1) mx = fmaxf(mx, __shfl_xor(mx, o, 32));
        float se = expf(xm - mx);
        for (int o = 16; o > 0; o >>= 1) se += __shfl_xor(se, o, 32);
        const float lp = xm - mx - logf(se);

        const float tclean = valid ? tl : 0.f;
        const bool  pos = tclean > 0.5f;
        float slp = pos ? lp : 0.f;
        float np  = pos ? 1.f : 0.f;
        for (int o = 16; o > 0; o >>= 1) {
            slp += __shfl_xor(slp, o, 32);
            np  += __shfl_xor(np,  o, 32);
        }

        // BCE
        const float scale = expf(logit_scale[0]) + 1e-9f;
        const float bsc   = fmaxf(fabsf(bce_scale[0]), 0.1f);
        const float x     = (v / scale) * bsc;
        const float per   = fmaxf(x, 0.f) - x * tl + log1pf(expf(-fabsf(x)));
        float bsum = valid ? per : 0.f;
        float bc   = valid ? 1.f : 0.f;
        for (int o = 16; o > 0; o >>= 1) {
            bsum += __shfl_xor(bsum, o, 32);
            bc   += __shfl_xor(bc,   o, 32);
        }

        if (t == 0) {
            const bool has = np > 0.f;
            ws_sup_sum[b] = has ? (-slp / (np + 1e-9f)) : 0.f;
            ws_anchor[b]  = has ? 1u : 0u;
            ws_bce_sum[b] = bsum;
            ws_bce_cnt[b] = (unsigned)(bc + 0.5f);
        }
    }
}

// Deterministic tree reduction over the 256 per-block partials + weighted combine.
__global__ __launch_bounds__(256)
void gliz_finalize(const float* __restrict__ rep_s, const float* __restrict__ sup_s,
                   const float* __restrict__ bce_s, const unsigned* __restrict__ rep_c,
                   const unsigned* __restrict__ anc, const unsigned* __restrict__ bce_c,
                   float* __restrict__ out)
{
    __shared__ float    fs[3][256];
    __shared__ unsigned us[3][256];
    const int t = threadIdx.x;
    fs[0][t] = rep_s[t]; fs[1][t] = sup_s[t]; fs[2][t] = bce_s[t];
    us[0][t] = rep_c[t]; us[1][t] = anc[t];   us[2][t] = bce_c[t];
    __syncthreads();
    for (int o = 128; o > 0; o >>= 1) {
        if (t < o) {
            fs[0][t] += fs[0][t + o];
            fs[1][t] += fs[1][t + o];
            fs[2][t] += fs[2][t + o];
            us[0][t] += us[0][t + o];
            us[1][t] += us[1][t + o];
            us[2][t] += us[2][t + o];
        }
        __syncthreads();
    }
    if (t == 0) {
        const unsigned rc = us[0][0] ? us[0][0] : 1u;
        const unsigned ac = us[1][0] ? us[1][0] : 1u;
        const unsigned bc = us[2][0] ? us[2][0] : 1u;
        const float supcon = fs[1][0] / (float)ac;
        const float repuls = fs[0][0] / (float)rc;
        const float bce    = fs[2][0] / (float)bc;
        out[0] = 1.0f * supcon + 0.1f * repuls + 1.0f * bce;
    }
}

extern "C" void kernel_launch(void* const* d_in, const int* in_sizes, int n_in,
                              void* d_out, int out_size, void* d_ws, size_t ws_size,
                              hipStream_t stream) {
    const float*     logits  = (const float*)d_in[0];
    const int*       labels  = (const int*)d_in[1];
    const long long* bidx    = (const long long*)d_in[2];
    const long long* lids    = (const long long*)d_in[3];
    const float*     emb     = (const float*)d_in[4];
    const float*     lsc     = (const float*)d_in[5];
    const float*     bsc     = (const float*)d_in[6];

    float* ws = (float*)d_ws;
    float* rep_s = ws;
    float* sup_s = ws + BB;
    float* bce_s = ws + 2 * BB;
    unsigned* rep_c = (unsigned*)(ws + 3 * BB);
    unsigned* anc   = rep_c + BB;
    unsigned* bce_c = rep_c + 2 * BB;

    gliz_block_kernel<<<BB, 128, SMEM_BYTES, stream>>>(logits, labels, bidx, lids,
                                                       emb, lsc, bsc,
                                                       rep_s, sup_s, bce_s,
                                                       rep_c, anc, bce_c);
    gliz_finalize<<<1, 256, 0, stream>>>(rep_s, sup_s, bce_s, rep_c, anc, bce_c,
                                         (float*)d_out);
}
